// GnDecoder_45509473468814
// MI455X (gfx1250) — matrix operand order repaired and checked by reading, not compiled, not run on hardware
//
#include <hip/hip_runtime.h>

typedef __attribute__((ext_vector_type(16))) _Float16 v16h;
typedef __attribute__((ext_vector_type(8)))  float    v8f;
typedef __attribute__((ext_vector_type(4)))  float    v4f;   // native vector for NT stores

#define NWAVES 8
#define TPAD   68            // padded LDS row stride (floats) to dodge bank conflicts
#define WPACK_ELEMS  36864   // total f16 weight elements (all 8 layers, fragment order)
#define PPACK_FLOATS 768     // 8 biases + gamma0/beta0/gamma1/beta1, 64 floats each

struct PackArgs {
  const float* W[8];   // weight matrices, (fan_in x 64) row-major f32
  const float* Bv[8];  // biases
  const float* G[2];   // gamma0, gamma1
  const float* Be[2];  // beta0, beta1
};

// ---------------------------------------------------------------------------
// Pack weights into v_wmma_f32_16x16x32_f16 B-fragment order (f16), and copy
// the small per-layer vectors (bias/gamma/beta) as f32.
// Fragment layout (B = 32x16, KxN): lane half h = lane>>4, elem e -> K = 8h + e (e<8),
// K = 16 + 8h + (e-8) (e>=8); N = lane&15. Frag = 32 lanes * 16 f16 = 512 f16.
// Layer frag order: (nt, kt) row-major -> frag index nt*Kt + kt.
// Elem base per layer: L0..L3: L*4096; L4(K=128): 16384; L5..L7: 24576+(L-5)*4096.
// ---------------------------------------------------------------------------
__global__ void pack_params_kernel(PackArgs pa, _Float16* __restrict__ wpack,
                                   float* __restrict__ ppack) {
  int tid = blockIdx.x * 256 + threadIdx.x;
  if (tid < WPACK_ELEMS) {
    int frag = tid >> 9;
    int r    = tid & 511;
    int lane = r >> 4;
    int e    = r & 15;
    int layer, fil, Kt;
    if (frag < 32)      { layer = frag >> 3;            fil = frag & 7;      Kt = 2; }
    else if (frag < 48) { layer = 4;                    fil = frag - 32;     Kt = 4; }
    else                { layer = 5 + ((frag - 48) >> 3); fil = (frag - 48) & 7; Kt = 2; }
    int nt = fil / Kt;
    int kt = fil % Kt;
    int k  = kt * 32 + (e & 7) + ((lane >> 4) << 3) + ((e >> 3) << 4);
    int n  = nt * 16 + (lane & 15);
    wpack[tid] = (_Float16)pa.W[layer][k * 64 + n];
  } else if (tid < WPACK_ELEMS + PPACK_FLOATS) {
    int o = tid - WPACK_ELEMS;
    int j = o >> 6, c = o & 63;
    const float* src;
    if      (j <  8) src = pa.Bv[j];
    else if (j == 8) src = pa.G[0];
    else if (j == 9) src = pa.Be[0];
    else if (j == 10) src = pa.G[1];
    else              src = pa.Be[1];
    ppack[o] = src[c];
  }
}

// Build a 16x32 f16 A-fragment per the CDNA5 ISA layout:
// lane m = lane&15 holds row m; elems 0..7 = K[k0..k0+7], elems 8..15 = K[k0+16..k0+23]
// where k0 = 8*(lane>>4). Caller passes pointers to the two 8-float runs.
__device__ __forceinline__ v16h make_a(const float* r0, const float* r1) {
  float4 x0 = *(const float4*)r0;
  float4 x1 = *(const float4*)(r0 + 4);
  float4 y0 = *(const float4*)r1;
  float4 y1 = *(const float4*)(r1 + 4);
  v16h a;
  a[0]  = (_Float16)x0.x; a[1]  = (_Float16)x0.y; a[2]  = (_Float16)x0.z; a[3]  = (_Float16)x0.w;
  a[4]  = (_Float16)x1.x; a[5]  = (_Float16)x1.y; a[6]  = (_Float16)x1.z; a[7]  = (_Float16)x1.w;
  a[8]  = (_Float16)y0.x; a[9]  = (_Float16)y0.y; a[10] = (_Float16)y0.z; a[11] = (_Float16)y0.w;
  a[12] = (_Float16)y1.x; a[13] = (_Float16)y1.y; a[14] = (_Float16)y1.z; a[15] = (_Float16)y1.w;
  return a;
}

__device__ __forceinline__ v8f wmma32(v16h a, v16h b, v8f c) {
  // D = A(16x32) * B(32x16) + C, f32 accumulate
  return __builtin_amdgcn_wmma_f32_16x16x32_f16(false, a, false, b, (short)0, c, false, false);
}

// ---------------------------------------------------------------------------
// Shared tail: 3 dense 64->64 layers (ReLU on first two) from an activation
// tile already staged in this wave's LDS tile, then LayerNorm + writeback.
// NT = true issues the final stores non-temporally (streaming output that
// must not evict L2-resident x_node1).
// ---------------------------------------------------------------------------
template <bool NT>
__device__ __forceinline__ void dense_layers_ln(
    float (*tile)[TPAD], float* mu, float* rs,
    const _Float16* __restrict__ wpack, int wb0, int wb1, int wb2,
    const float* __restrict__ pp, int bi0,
    const float* __restrict__ gamma, const float* __restrict__ beta,
    float* __restrict__ out, int lane) {
  const int m  = lane & 15;
  const int hi = lane >> 4;
  const int wb[3] = {wb0, wb1, wb2};
#pragma unroll
  for (int L = 0; L < 3; ++L) {
    v16h a0 = make_a(&tile[m][hi * 8],      &tile[m][hi * 8 + 16]);
    v16h a1 = make_a(&tile[m][32 + hi * 8], &tile[m][32 + hi * 8 + 16]);
    __syncthreads();   // reads done before the tile is overwritten
    const bool  relu = (L < 2);
    const float* bl  = pp + (bi0 + L) * 64;
#pragma unroll
    for (int nt = 0; nt < 4; ++nt) {
      float bn = bl[nt * 16 + m];
      v8f c = {bn, bn, bn, bn, bn, bn, bn, bn};
      v16h w0 = *(const v16h*)(wpack + wb[L] + (nt * 2 + 0) * 512 + lane * 16);
      v16h w1 = *(const v16h*)(wpack + wb[L] + (nt * 2 + 1) * 512 + lane * 16);
      c = wmma32(a0, w0, c);
      c = wmma32(a1, w1, c);
#pragma unroll
      for (int r = 0; r < 8; ++r) {
        float v = c[r];
        if (relu) v = v > 0.f ? v : 0.f;
        tile[r + hi * 8][nt * 16 + m] = v;   // C layout: M = r + 8*hi, N = nt*16 + m
      }
    }
    __syncthreads();
  }
  // LayerNorm over 64 features per row
  if (lane < 16) {
    float s = 0.f, s2 = 0.f;
#pragma unroll
    for (int c4 = 0; c4 < 16; ++c4) {
      float4 v = *(const float4*)&tile[lane][c4 * 4];
      s  += v.x + v.y + v.z + v.w;
      s2 += v.x * v.x + v.y * v.y + v.z * v.z + v.w * v.w;
    }
    float mean = s * (1.f / 64.f);
    float var  = s2 * (1.f / 64.f) - mean * mean;
    mu[lane] = mean;
    rs[lane] = rsqrtf(var + 1e-5f);
  }
  __syncthreads();
#pragma unroll
  for (int i = 0; i < 8; ++i) {
    int idx4 = i * 32 + lane;          // 256 float4 = 16 rows x 16 col-groups
    int row = idx4 >> 4, c4 = idx4 & 15;
    float4 v  = *(const float4*)&tile[row][c4 * 4];
    float  mm = mu[row], rr = rs[row];
    float4 g  = *(const float4*)(gamma + c4 * 4);
    float4 be = *(const float4*)(beta  + c4 * 4);
    v4f o;
    o[0] = (v.x - mm) * rr * g.x + be.x;
    o[1] = (v.y - mm) * rr * g.y + be.y;
    o[2] = (v.z - mm) * rr * g.z + be.z;
    o[3] = (v.w - mm) * rr * g.w + be.w;
    v4f* dst = (v4f*)(out + row * 64 + c4 * 4);
    if (NT) __builtin_nontemporal_store(o, dst);   // TH=NT streaming store
    else    *dst = o;
  }
}

// ---------------------------------------------------------------------------
// Node MLP: x_node1 = mlp_ln(x_node, params0). One wave per 16 nodes.
// Output kept regular-temporal: it must stay L2-resident for the edge gather.
// ---------------------------------------------------------------------------
__global__ void __launch_bounds__(256) node_mlp(
    const float* __restrict__ x, const _Float16* __restrict__ wpack,
    const float* __restrict__ pp, float* __restrict__ out, int ntiles) {
  __shared__ __align__(16) float tile[NWAVES][16][TPAD];
  __shared__ float muS[NWAVES][16], rsS[NWAVES][16];
  const int wave = threadIdx.x >> 5;
  const int lane = threadIdx.x & 31;
  int t = blockIdx.x * NWAVES + wave;
  if (t >= ntiles) t = ntiles - 1;          // duplicate work, identical writes
  const int m = lane & 15, hi = lane >> 4;

  const float* row = x + (size_t)(t * 16 + m) * 64;
  v16h a0 = make_a(row + hi * 8,      row + hi * 8 + 16);
  v16h a1 = make_a(row + 32 + hi * 8, row + 32 + hi * 8 + 16);
  const float* bl = pp;                     // bias of layer 0
#pragma unroll
  for (int nt = 0; nt < 4; ++nt) {
    float bn = bl[nt * 16 + m];
    v8f c = {bn, bn, bn, bn, bn, bn, bn, bn};
    v16h w0 = *(const v16h*)(wpack + (nt * 2 + 0) * 512 + lane * 16);
    v16h w1 = *(const v16h*)(wpack + (nt * 2 + 1) * 512 + lane * 16);
    c = wmma32(a0, w0, c);
    c = wmma32(a1, w1, c);
#pragma unroll
    for (int r = 0; r < 8; ++r) {
      float v = c[r];
      tile[wave][r + hi * 8][nt * 16 + m] = v > 0.f ? v : 0.f;  // ReLU
    }
  }
  __syncthreads();
  dense_layers_ln<false>(tile[wave], muS[wave], rsS[wave], wpack,
                         4096, 8192, 12288, pp, 1, pp + 512, pp + 576,
                         out + (size_t)t * 1024, lane);
}

// ---------------------------------------------------------------------------
// Edge MLP: gather sender/receiver rows of x_node1 (L2-resident), concat via
// K-tile selection (K=0..63 sender, 64..127 receiver), 4-layer MLP + LN.
// One wave per 16 edges. Output stored non-temporally (256 MB stream).
// ---------------------------------------------------------------------------
__global__ void __launch_bounds__(256) edge_mlp(
    const float* __restrict__ xn1, const int* __restrict__ eidx,
    const _Float16* __restrict__ wpack, const float* __restrict__ pp,
    float* __restrict__ out, int ntiles) {
  __shared__ __align__(16) float tile[NWAVES][16][TPAD];
  __shared__ float muS[NWAVES][16], rsS[NWAVES][16];
  const int wave = threadIdx.x >> 5;
  const int lane = threadIdx.x & 31;
  int t = blockIdx.x * NWAVES + wave;
  if (t >= ntiles) t = ntiles - 1;
  const int m = lane & 15, hi = lane >> 4;

  const int e  = t * 16 + m;
  const int si = __builtin_nontemporal_load(eidx + 2 * e);      // read-once stream
  const int ri = __builtin_nontemporal_load(eidx + 2 * e + 1);
  const float* rowS = xn1 + (size_t)si * 64;
  const float* rowR = xn1 + (size_t)ri * 64;

  v16h a[4];
  a[0] = make_a(rowS + hi * 8,      rowS + hi * 8 + 16);        // K   0..31 (sender)
  a[1] = make_a(rowS + 32 + hi * 8, rowS + 32 + hi * 8 + 16);   // K  32..63
  a[2] = make_a(rowR + hi * 8,      rowR + hi * 8 + 16);        // K  64..95 (receiver)
  a[3] = make_a(rowR + 32 + hi * 8, rowR + 32 + hi * 8 + 16);   // K 96..127

  const float* bl = pp + 4 * 64;            // bias of edge layer 0
#pragma unroll
  for (int nt = 0; nt < 4; ++nt) {
    float bn = bl[nt * 16 + m];
    v8f c = {bn, bn, bn, bn, bn, bn, bn, bn};
#pragma unroll
    for (int kt = 0; kt < 4; ++kt) {
      v16h w = *(const v16h*)(wpack + 16384 + (nt * 4 + kt) * 512 + lane * 16);
      c = wmma32(a[kt], w, c);
    }
#pragma unroll
    for (int r = 0; r < 8; ++r) {
      float v = c[r];
      tile[wave][r + hi * 8][nt * 16 + m] = v > 0.f ? v : 0.f;  // ReLU
    }
  }
  __syncthreads();
  dense_layers_ln<true>(tile[wave], muS[wave], rsS[wave], wpack,
                        24576, 28672, 32768, pp, 5, pp + 640, pp + 704,
                        out + (size_t)t * 1024, lane);
}

// ---------------------------------------------------------------------------
extern "C" void kernel_launch(void* const* d_in, const int* in_sizes, int n_in,
                              void* d_out, int out_size, void* d_ws, size_t ws_size,
                              hipStream_t stream) {
  const float* x = (const float*)d_in[0];
  PackArgs pa;
  // Two plausible pytree flatten orders for the params dicts:
  //  - sorted keys (jax tree_flatten):  beta, gamma, (W,b)x4
  //  - insertion order:                 (W,b)x4, gamma, beta
  if (in_sizes[1] == 64) {  // sorted: d_in[1]=beta0 (64)
    pa.Be[0] = (const float*)d_in[1];
    pa.G[0]  = (const float*)d_in[2];
    for (int i = 0; i < 4; ++i) {
      pa.W[i]  = (const float*)d_in[3 + 2 * i];
      pa.Bv[i] = (const float*)d_in[4 + 2 * i];
    }
    pa.Be[1] = (const float*)d_in[11];
    pa.G[1]  = (const float*)d_in[12];
    for (int i = 0; i < 4; ++i) {
      pa.W[4 + i]  = (const float*)d_in[13 + 2 * i];
      pa.Bv[4 + i] = (const float*)d_in[14 + 2 * i];
    }
  } else {                  // insertion: d_in[1]=W0 (4096)
    for (int i = 0; i < 4; ++i) {
      pa.W[i]  = (const float*)d_in[1 + 2 * i];
      pa.Bv[i] = (const float*)d_in[2 + 2 * i];
    }
    pa.G[0]  = (const float*)d_in[9];
    pa.Be[0] = (const float*)d_in[10];
    for (int i = 0; i < 4; ++i) {
      pa.W[4 + i]  = (const float*)d_in[11 + 2 * i];
      pa.Bv[4 + i] = (const float*)d_in[12 + 2 * i];
    }
    pa.G[1]  = (const float*)d_in[19];
    pa.Be[1] = (const float*)d_in[20];
  }
  const int* eidx = (const int*)d_in[n_in - 1];

  _Float16* wpack = (_Float16*)d_ws;
  float*    ppack = (float*)((char*)d_ws + (size_t)WPACK_ELEMS * 2);

  float* out = (float*)d_out;
  const int nodeTiles = 100000 / 16;   // 6250
  const int edgeTiles = 1000000 / 16;  // 62500

  pack_params_kernel<<<(WPACK_ELEMS + PPACK_FLOATS + 255) / 256, 256, 0, stream>>>(
      pa, wpack, ppack);
  node_mlp<<<(nodeTiles + NWAVES - 1) / NWAVES, 256, 0, stream>>>(
      x, wpack, ppack, out, nodeTiles);
  edge_mlp<<<(edgeTiles + NWAVES - 1) / NWAVES, 256, 0, stream>>>(
      out, eidx, wpack, ppack, out + (size_t)100000 * 64, edgeTiles);
}
